// MambaKANEncoder_25812753449332
// MI455X (gfx1250) — compile-verified, matrix operand order checked by reading, not run on hardware
//
#include <hip/hip_runtime.h>
#include <hip/hip_bf16.h>
#include <cstdint>

// ---- model constants (from the reference) ----
#define NB_MOL   256
#define HIDDEN_  128
#define IN_DIM_  64
#define D_INNER_ 256
#define D_STATE_ 16
#define DT_RANK_ 8
#define D_CONV_  4
#define XDBL_LD  64          // xdbl row stride, padded 40 -> 64

// ---- GEMM tiling ----
#define TM 128               // M rows per workgroup
#define TN 64                // N cols per workgroup
#define TK 32                // K per step (one f16 WMMA depth)
#define LPITCH 40            // LDS pitch in halfs (80B rows, 16B aligned, staggers banks)

typedef __attribute__((ext_vector_type(16))) _Float16 v16h;
typedef __attribute__((ext_vector_type(8)))  _Float16 v8h;
typedef __attribute__((ext_vector_type(4)))  _Float16 v4h;
typedef __attribute__((ext_vector_type(8)))  float    v8f;

union F16x16 { v16h v; v8h h[2]; };

// -------------------------------------------------------------------------
// starts[b] = first packed row of molecule b (batch is sorted); starts[256]=N
// -------------------------------------------------------------------------
__global__ void compute_starts_kernel(const int* __restrict__ batch, int Ntok,
                                      int* __restrict__ starts) {
  int b = blockIdx.x * blockDim.x + threadIdx.x;
  if (b > NB_MOL) return;
  if (b == NB_MOL) { starts[b] = Ntok; return; }
  int lo = 0, hi = Ntok;
  while (lo < hi) {
    int mid = (lo + hi) >> 1;
    if (batch[mid] < b) lo = mid + 1; else hi = mid;
  }
  starts[b] = lo;
}

// -------------------------------------------------------------------------
// Tiled f16-WMMA GEMM, no bounds checks (all dims pre-padded):
//   C[M,Nn] = A[M,K](lda) * B[K,Nn](ldb) (+ bias) (+ C if accumulate)
// Block = 256 threads = 8 waves; 128x64 tile.
// Software-pipelined double-buffered LDS: global loads for tile k+1 are in
// flight (6 independent b128 loads) while tile k's 4 WMMAs execute; one
// workgroup barrier per k-step. Wave w owns rows [16w,16w+16); per k-step:
// 1 A-frag + 4 B-frags (2x ds_load_b128 each), then 4 back-to-back WMMAs.
// Fragment layouts per CDNA5 ISA 7.12.2 (lane[3:0]=row/col, lane[4]=K-half).
// -------------------------------------------------------------------------
__global__ __launch_bounds__(256)
void wmma_gemm_tiled(const float* __restrict__ A, int lda,
                     const float* __restrict__ Bw, int ldb,
                     const float* __restrict__ bias,
                     float* __restrict__ C, int ldc,
                     int K, int accumulate)
{
  __shared__ _Float16 As[2][TM][LPITCH];   // [buf][row][k]
  __shared__ _Float16 Bs[2][TN][LPITCH];   // [buf][col][k] (transposed)

  const int t    = threadIdx.x;
  const int lane = t & 31;
  const int w    = t >> 5;                  // wave id 0..7
  const int m0   = blockIdx.x * TM;
  const int n0   = blockIdx.y * TN;

  // staging coordinates
  const int arow  = t >> 3;                 // 0..31  (A: 4 row-passes of 32)
  const int acol4 = (t & 7) * 4;            // 0..28
  const int bk    = t >> 4;                 // 0..15  (B: 2 k-passes of 16)
  const int bc4   = (t & 15) * 4;           // 0..60

  // fragment coordinates
  const int frow  = w * 16 + (lane & 15);
  const int khalf = (lane >> 4) * 8;

  float4 ar[4], br[2];                      // in-flight staging registers

  auto load_regs = [&](int k0) {
#pragma unroll
    for (int p = 0; p < 4; ++p)
      ar[p] = *(const float4*)&A[(size_t)(m0 + arow + p * 32) * lda + k0 + acol4];
#pragma unroll
    for (int p = 0; p < 2; ++p)
      br[p] = *(const float4*)&Bw[(size_t)(k0 + bk + p * 16) * ldb + n0 + bc4];
  };

  auto convert_store = [&](int buf) {
#pragma unroll
    for (int p = 0; p < 4; ++p) {
      v4h hv = { (_Float16)ar[p].x, (_Float16)ar[p].y,
                 (_Float16)ar[p].z, (_Float16)ar[p].w };
      *(v4h*)&As[buf][arow + p * 32][acol4] = hv;
    }
#pragma unroll
    for (int p = 0; p < 2; ++p) {
      int k = bk + p * 16;
      Bs[buf][bc4 + 0][k] = (_Float16)br[p].x;
      Bs[buf][bc4 + 1][k] = (_Float16)br[p].y;
      Bs[buf][bc4 + 2][k] = (_Float16)br[p].z;
      Bs[buf][bc4 + 3][k] = (_Float16)br[p].w;
    }
  };

  v8f acc[4] = {};
  load_regs(0);
  int buf = 0;
  for (int k0 = 0; k0 < K; k0 += TK) {
    convert_store(buf);                 // store tile k0 (regs -> LDS[buf])
    if (k0 + TK < K) load_regs(k0 + TK);// issue next tile's 6 b128 loads now
    __syncthreads();                    // LDS[buf] ready for all waves

    F16x16 a, b[4];
    a.h[0] = *(const v8h*)&As[buf][frow][khalf];
    a.h[1] = *(const v8h*)&As[buf][frow][16 + khalf];
#pragma unroll
    for (int ns = 0; ns < 4; ++ns) {
      const int bcol = ns * 16 + (lane & 15);
      b[ns].h[0] = *(const v8h*)&Bs[buf][bcol][khalf];
      b[ns].h[1] = *(const v8h*)&Bs[buf][bcol][16 + khalf];
    }
#pragma unroll
    for (int ns = 0; ns < 4; ++ns)      // 4 WMMAs back-to-back
      acc[ns] = __builtin_amdgcn_wmma_f32_16x16x32_f16(
          false, a.v, false, b[ns].v, (short)0, acc[ns], false, false);
    buf ^= 1;
  }

  // ---- epilogue: unguarded (padded) stores ----
#pragma unroll
  for (int ns = 0; ns < 4; ++ns) {
#pragma unroll
    for (int r = 0; r < 8; ++r) {
      int m = m0 + w * 16 + r + ((lane >> 4) << 3);
      int n = n0 + ns * 16 + (lane & 15);
      float v = acc[ns][r];
      if (bias)       v += bias[n];
      if (accumulate) v += C[(size_t)m * ldc + n];
      C[(size_t)m * ldc + n] = v;
    }
  }
}

// -------------------------------------------------------------------------
// Copy x into M-padded buffer (pad rows = 0).
// -------------------------------------------------------------------------
__global__ __launch_bounds__(IN_DIM_)
void pad_x_kernel(const float* __restrict__ x, float* __restrict__ xpad, int Ntok) {
  int r = blockIdx.x, c = threadIdx.x;
  xpad[(size_t)r * IN_DIM_ + c] = (r < Ntok) ? x[(size_t)r * IN_DIM_ + c] : 0.0f;
}

// Pad xprojW [256,40] -> [256,64] (zero cols 40..63).
__global__ __launch_bounds__(XDBL_LD)
void pad_wproj_kernel(const float* __restrict__ W, float* __restrict__ Wp) {
  int k = blockIdx.x, n = threadIdx.x;
  Wp[k * XDBL_LD + n] = (n < DT_RANK_ + 2 * D_STATE_) ? W[k * (DT_RANK_ + 2 * D_STATE_) + n] : 0.0f;
}

// Extract dtr (cols 0..7 of xdbl) into K=32 zero-padded operand.
__global__ __launch_bounds__(32)
void dtr_pad_kernel(const float* __restrict__ xdbl, float* __restrict__ dtr) {
  int r = blockIdx.x, c = threadIdx.x;
  dtr[(size_t)r * 32 + c] = (c < DT_RANK_) ? xdbl[(size_t)r * XDBL_LD + c] : 0.0f;
}

// -------------------------------------------------------------------------
// RMSNorm over HIDDEN=128 features. One block per (padded) row.
// -------------------------------------------------------------------------
__global__ __launch_bounds__(HIDDEN_)
void rmsnorm_kernel(const float* __restrict__ h, const float* __restrict__ w,
                    float* __restrict__ xn) {
  __shared__ float red[HIDDEN_];
  int r = blockIdx.x;
  int f = threadIdx.x;
  float v = h[(size_t)r * HIDDEN_ + f];
  red[f] = v * v;
  __syncthreads();
  for (int ofs = HIDDEN_ / 2; ofs > 0; ofs >>= 1) {
    if (f < ofs) red[f] += red[f + ofs];
    __syncthreads();
  }
  float rms = rsqrtf(red[0] * (1.0f / HIDDEN_) + 1e-6f);
  xn[(size_t)r * HIDDEN_ + f] = v * rms * w[f];
}

// -------------------------------------------------------------------------
// Causal depthwise conv (k=4) + bias + SiLU on the packed layout.
// Runs over padded rows; pad rows write 0 (keeps later GEMM inputs defined).
// -------------------------------------------------------------------------
__global__ __launch_bounds__(D_INNER_)
void conv_silu_kernel(const float* __restrict__ xz,
                      const int* __restrict__ batch,
                      const int* __restrict__ starts,
                      const float* __restrict__ convw,
                      const float* __restrict__ convb,
                      float* __restrict__ uc, int Ntok) {
  int r = blockIdx.x;
  int d = threadIdx.x;
  if (r >= Ntok) { uc[(size_t)r * D_INNER_ + d] = 0.0f; return; }
  int b = batch[r];
  int tpos = r - starts[b];
  float acc = convb[d];
#pragma unroll
  for (int j = 0; j < D_CONV_; ++j) {
    int tt = tpos - (D_CONV_ - 1) + j;
    if (tt >= 0)
      acc += convw[d * D_CONV_ + j] *
             xz[(size_t)(r - (D_CONV_ - 1) + j) * (2 * D_INNER_) + d];
  }
  acc = acc * (1.0f / (1.0f + __expf(-acc)));
  uc[(size_t)r * D_INNER_ + d] = acc;
}

// -------------------------------------------------------------------------
// Selective scan; 16 states in registers per (molecule, channel) lane.
// Fuses softplus(dt), u*Dskip, silu(z) gate; writes y over dead u-half of xz.
// -------------------------------------------------------------------------
__global__ __launch_bounds__(D_INNER_)
void scan_kernel(const float* __restrict__ uc,
                 const float* __restrict__ xdbl,   // [*, 64]: dtr|B|C|pad
                 const float* __restrict__ dtpre,  // [*, 256] (bias included)
                 const float* __restrict__ A_log,
                 const float* __restrict__ Dskip,
                 float* __restrict__ xz,
                 const int* __restrict__ starts) {
  int b = blockIdx.x;
  int d = threadIdx.x;
  int start = starts[b];
  int len   = starts[b + 1] - start;

  float Aneg[D_STATE_];
#pragma unroll
  for (int s = 0; s < D_STATE_; ++s)
    Aneg[s] = -__expf(A_log[d * D_STATE_ + s]);

  float hs[D_STATE_];
#pragma unroll
  for (int s = 0; s < D_STATE_; ++s) hs[s] = 0.0f;

  float dsk = Dskip[d];
  for (int tpos = 0; tpos < len; ++tpos) {
    size_t r = (size_t)(start + tpos);
    float xdt = dtpre[r * D_INNER_ + d];
    float dt  = (xdt > 20.0f) ? xdt : log1pf(__expf(xdt));
    float u   = uc[r * D_INNER_ + d];
    const float* xd = xdbl + r * XDBL_LD;
    float du = dt * u;
    float y  = 0.0f;
#pragma unroll
    for (int s = 0; s < D_STATE_; ++s) {
      float dA = __expf(dt * Aneg[s]);
      hs[s] = dA * hs[s] + du * xd[DT_RANK_ + s];
      y += hs[s] * xd[DT_RANK_ + D_STATE_ + s];
    }
    y += u * dsk;
    float z = xz[r * (2 * D_INNER_) + D_INNER_ + d];
    y *= z * (1.0f / (1.0f + __expf(-z)));
    xz[r * (2 * D_INNER_) + d] = y;
  }
}

// -------------------------------------------------------------------------
// Masked mean pool over each molecule's valid rows.
// -------------------------------------------------------------------------
__global__ __launch_bounds__(HIDDEN_)
void pool_kernel(const float* __restrict__ h, const int* __restrict__ starts,
                 float* __restrict__ out) {
  int b = blockIdx.x;
  int f = threadIdx.x;
  int start = starts[b];
  int len   = starts[b + 1] - start;
  float s = 0.0f;
  for (int tpos = 0; tpos < len; ++tpos)
    s += h[(size_t)(start + tpos) * HIDDEN_ + f];
  out[b * HIDDEN_ + f] = s / (float)(len > 0 ? len : 1);
}

// -------------------------------------------------------------------------
extern "C" void kernel_launch(void* const* d_in, const int* in_sizes, int n_in,
                              void* d_out, int out_size, void* d_ws, size_t ws_size,
                              hipStream_t stream) {
  const float* x      = (const float*)d_in[0];
  const int*   batch  = (const int*)  d_in[1];
  const float* in_w   = (const float*)d_in[2];
  const float* in_b   = (const float*)d_in[3];
  const float* norm_w = (const float*)d_in[4];
  const float* inW    = (const float*)d_in[5];
  const float* conv_w = (const float*)d_in[6];
  const float* conv_b = (const float*)d_in[7];
  const float* xprojW = (const float*)d_in[8];
  const float* dtW    = (const float*)d_in[9];
  const float* dt_b   = (const float*)d_in[10];
  const float* A_log  = (const float*)d_in[11];
  const float* Dskip  = (const float*)d_in[12];
  const float* outW   = (const float*)d_in[13];
  (void)n_in; (void)out_size; (void)ws_size;

  const int Ntok = in_sizes[1];                      // packed token count
  const int Mpad = ((Ntok + TM - 1) / TM) * TM;      // multiple of 128
  const int Mt   = Mpad / TM;                        // M tile count

  // ---- workspace carve-out (packed layout, M-padded; ~100 MB) ----
  char* wsp = (char*)d_ws;
  size_t off = 0;
  auto carve = [&](size_t bytes) -> void* {
    void* p = wsp + off;
    off += (bytes + 255) & ~(size_t)255;
    return p;
  };
  int*   starts  = (int*)  carve((NB_MOL + 1) * sizeof(int));
  float* xpad    = (float*)carve((size_t)Mpad * IN_DIM_      * sizeof(float));
  float* h       = (float*)carve((size_t)Mpad * HIDDEN_      * sizeof(float));
  float* xn      = (float*)carve((size_t)Mpad * HIDDEN_      * sizeof(float));
  float* xz      = (float*)carve((size_t)Mpad * 2 * D_INNER_ * sizeof(float));
  float* uc      = (float*)carve((size_t)Mpad * D_INNER_     * sizeof(float));
  float* xdbl    = (float*)carve((size_t)Mpad * XDBL_LD      * sizeof(float));
  float* dtrp    = (float*)carve((size_t)Mpad * 32           * sizeof(float));
  float* dtpre   = (float*)carve((size_t)Mpad * D_INNER_     * sizeof(float));
  float* wprojp  = (float*)carve((size_t)D_INNER_ * XDBL_LD  * sizeof(float));

  compute_starts_kernel<<<dim3((NB_MOL + 1 + 63) / 64), dim3(64), 0, stream>>>(
      batch, Ntok, starts);
  pad_x_kernel<<<Mpad, IN_DIM_, 0, stream>>>(x, xpad, Ntok);

  // h = xpad @ in_w + in_b     [Mpad,64]x[64,128]
  wmma_gemm_tiled<<<dim3(Mt, HIDDEN_ / TN), 256, 0, stream>>>(
      xpad, IN_DIM_, in_w, HIDDEN_, in_b, h, HIDDEN_, IN_DIM_, 0);

  for (int l = 0; l < 2; ++l) {
    rmsnorm_kernel<<<Mpad, HIDDEN_, 0, stream>>>(h, norm_w + l * HIDDEN_, xn);

    // xz = xn @ inW[l]         [Mpad,128]x[128,512]
    wmma_gemm_tiled<<<dim3(Mt, (2 * D_INNER_) / TN), 256, 0, stream>>>(
        xn, HIDDEN_, inW + (size_t)l * HIDDEN_ * 2 * D_INNER_, 2 * D_INNER_,
        nullptr, xz, 2 * D_INNER_, HIDDEN_, 0);

    conv_silu_kernel<<<Mpad, D_INNER_, 0, stream>>>(
        xz, batch, starts, conv_w + (size_t)l * D_INNER_ * D_CONV_,
        conv_b + l * D_INNER_, uc, Ntok);

    // xdbl = uc @ pad(xprojW[l])   [Mpad,256]x[256,64]
    pad_wproj_kernel<<<D_INNER_, XDBL_LD, 0, stream>>>(
        xprojW + (size_t)l * D_INNER_ * (DT_RANK_ + 2 * D_STATE_), wprojp);
    wmma_gemm_tiled<<<dim3(Mt, XDBL_LD / TN), 256, 0, stream>>>(
        uc, D_INNER_, wprojp, XDBL_LD, nullptr, xdbl, XDBL_LD, D_INNER_, 0);

    // dtpre = pad(dtr) @ dtW[l] + dt_b[l]   [Mpad,32]x[(8->32),256]
    dtr_pad_kernel<<<Mpad, 32, 0, stream>>>(xdbl, dtrp);
    wmma_gemm_tiled<<<dim3(Mt, D_INNER_ / TN), 256, 0, stream>>>(
        dtrp, 32, dtW + (size_t)l * DT_RANK_ * D_INNER_, D_INNER_,
        dt_b + l * D_INNER_, dtpre, D_INNER_, 32, 0);

    scan_kernel<<<NB_MOL, D_INNER_, 0, stream>>>(
        uc, xdbl, dtpre, A_log + (size_t)l * D_INNER_ * D_STATE_,
        Dskip + l * D_INNER_, xz, starts);

    // h += y @ outW[l]          [Mpad,256]x[256,128]
    wmma_gemm_tiled<<<dim3(Mt, HIDDEN_ / TN), 256, 0, stream>>>(
        xz, 2 * D_INNER_, outW + (size_t)l * D_INNER_ * HIDDEN_, HIDDEN_,
        nullptr, h, HIDDEN_, D_INNER_, 1);
  }

  pool_kernel<<<NB_MOL, HIDDEN_, 0, stream>>>(h, starts, (float*)d_out);
}
